// VSDN_VAE_SMOOTH_32933809225760
// MI455X (gfx1250) — matrix operand order repaired and checked by reading, not compiled, not run on hardware
//
#include <hip/hip_runtime.h>
#include <math.h>

// ---------------- problem constants ----------------
#define T_STEPS 128
#define B_DIM   128
#define S_DIM   8
#define DIN     64
#define R_DIM   256
#define Z_DIM   128
#define H_DIM   512
#define KD      (R_DIM + Z_DIM)      // 384
#define MT      32                   // (b,s) rows per workgroup
#define NWG     (B_DIM * S_DIM / MT) // 32 workgroups
#define NTHREADS 256                 // 8 waves of 32
#define ACTS    392                  // act LDS row stride (bf16 elems), even
#define A1S     520                  // a1  LDS row stride (bf16 elems), even

#define ZOUT    (B_DIM * S_DIM * Z_DIM) // 131072 floats of z in d_out
#define LOG2PI  1.8378770664093453f

// swizzled-weight offsets in workspace (bf16 elements; all 16-elem aligned)
#define WD1_OFF 0                        // 384*512
#define WD2_OFF 196608                   // 512*128
#define WF1_OFF 262144                   // 256*512
#define WF2_OFF 393216                   // 512*128
#define WP1_OFF 458752                   // 384*512
#define WP2_OFF 655360                   // 512*128

typedef __attribute__((ext_vector_type(16))) __bf16 bf16x16;
typedef __attribute__((ext_vector_type(8)))  float  f32x8;

union ABf { bf16x16 v; unsigned int u[8]; uint4 q[2]; };

// ---------------- prep: f32 weight [K,N] -> bf16 swizzled B-fragments ----------------
// Per 32x16 (KxN) tile, lane L: n = nt*16 + (L&15), k = kt*32 + 16*(L>>4) + {0..15}
// stored contiguously (16 bf16 = 32B per lane -> two b128 loads).
__global__ void swizzle_weights(const float* __restrict__ W,
                                __bf16* __restrict__ dst, int K, int N) {
  int idx = blockIdx.x * blockDim.x + threadIdx.x;
  int total = K * N;
  if (idx >= total) return;
  int NT = N >> 4;
  int tile = idx >> 9, rem = idx & 511;
  int lane = rem >> 4, v2 = rem & 15;
  int ktile = tile / NT, ntile = tile - ktile * NT;
  int k = (ktile << 5) + ((lane >> 4) << 4) + v2;
  int n = (ntile << 4) + (lane & 15);
  dst[idx] = (__bf16)W[(size_t)k * N + n];   // native RNE f32->bf16 cvt
}

// ---------------- WMMA fragment helpers ----------------
// A-matrix 16x32 bf16: lane L row m=L&15, half=L>>4;
// VGPR0..3: K = 16*half + {0..7}; VGPR4..7: K = 16*half + 8 + {0..7}.
__device__ __forceinline__ void load_afrag(const __bf16* act, int stride,
                                           int kbase, int lane, ABf& a) {
  int m = lane & 15, half = lane >> 4;
  const unsigned short* row =
      (const unsigned short*)(act + m * stride + kbase + half * 16);
#pragma unroll
  for (int v = 0; v < 4; ++v) a.u[v] = *(const unsigned int*)(row + 2 * v);
#pragma unroll
  for (int v = 0; v < 4; ++v) a.u[4 + v] = *(const unsigned int*)(row + 8 + 2 * v);
}

__device__ __forceinline__ void load_bfrag(const __bf16* __restrict__ Wsw,
                                           int NT, int ktile, int ntile, int lane, ABf& b) {
  const uint4* q = (const uint4*)(Wsw + (((size_t)(ktile * NT + ntile) * 32 + lane) << 4));
  b.q[0] = q[0];
  b.q[1] = q[1];
}

// NMT vertically-stacked 16x16 output tiles sharing each B fragment (register M-blocking)
template <int NMT>
__device__ __forceinline__ void gemm_multi(const __bf16* act, int actStride, int K,
                                           const __bf16* __restrict__ Wsw, int NT,
                                           int ntile, const float* __restrict__ bias,
                                           int lane, f32x8* acc) {
  float bv = bias[(ntile << 4) + (lane & 15)];
#pragma unroll
  for (int mt = 0; mt < NMT; ++mt)
#pragma unroll
    for (int r = 0; r < 8; ++r) acc[mt][r] = bv;
  const int KT = K >> 5;
#pragma unroll 2
  for (int kt = 0; kt < KT; ++kt) {
    ABf b;
    load_bfrag(Wsw, NT, kt, ntile, lane, b);
#pragma unroll
    for (int mt = 0; mt < NMT; ++mt) {
      ABf a;
      load_afrag(act + mt * 16 * actStride, actStride, kt << 5, lane, a);
      acc[mt] = __builtin_amdgcn_wmma_f32_16x16x32_bf16(false, a.v, false, b.v,
                                                        (short)0, acc[mt], false, false);
    }
  }
}

// ---------------- scalar init / epilogue ----------------
__global__ void zero_scalars(float* out) {
  if (threadIdx.x < 3) out[ZOUT + threadIdx.x] = 0.0f;
}
__global__ void finalize_total(float* out) {
  if (threadIdx.x == 0) out[ZOUT] = out[ZOUT + 1] + out[ZOUT + 2];
}

// ---------------- main persistent-scan kernel ----------------
// Each WG owns 32 (b,s)-rows (4 batch rows x 8 samples), keeps z in LDS across all
// 128 steps.  prior & poster fused as a single 64-row GEMM through Wd1/Wd2.
__global__ __launch_bounds__(NTHREADS) void vsdn_kernel(
    const float* __restrict__ X, const float* __restrict__ Mmask,
    const float* __restrict__ cov, const float* __restrict__ path_h,
    const float* __restrict__ path_hpos, const float* __restrict__ noise,
    const float* __restrict__ Wc1, const float* __restrict__ bc1,
    const float* __restrict__ Wc2, const float* __restrict__ bc2,
    const float* __restrict__ bd1, const float* __restrict__ bd2,
    const float* __restrict__ bf1, const float* __restrict__ bf2,
    const float* __restrict__ bp1, const float* __restrict__ bp2,
    const __bf16* __restrict__ ws, float* __restrict__ out) {
  __shared__ __bf16 act[2 * MT * ACTS]; // rows 0..31:[h|z], rows 32..63:[h+hpos|z]
  __shared__ __bf16 a1[2 * MT * A1S];   // hidden activations (bf16)
  __shared__ float ddb[2 * MT * Z_DIM]; // rows 0..31: prior, 32..63: poster
  __shared__ float diffb[MT * Z_DIM];
  __shared__ float pbuf[MT * Z_DIM];
  __shared__ float zbuf[MT * Z_DIM];
  __shared__ float hid[4 * 64];
  __shared__ float red[NTHREADS];
  __shared__ float MtS[4];

  const int tid = threadIdx.x;
  const int lane = tid & 31, wave = tid >> 5;
  const int wg = blockIdx.x;
  const int m0 = wg * MT;   // global (b,s)-row base
  const int b0 = m0 >> 3;   // four unique batch rows per WG

  const __bf16* Wd1s = ws + WD1_OFF;
  const __bf16* Wd2s = ws + WD2_OFF;
  const __bf16* Wf1s = ws + WF1_OFF;
  const __bf16* Wf2s = ws + WF2_OFF;
  const __bf16* Wp1s = ws + WP1_OFF;
  const __bf16* Wp2s = ws + WP2_OFF;

  // ---- z0 = tanh(relu(cov@Wc1+bc1)@Wc2+bc2), broadcast over S ----
  {
    int bi = tid >> 6, j = tid & 63;  // 4 x 64 = 256 threads
    const float* cr = cov + (size_t)(b0 + bi) * 32;
    float s = bc1[j];
#pragma unroll 8
    for (int c = 0; c < 32; ++c) s += cr[c] * Wc1[c * 64 + j];
    hid[bi * 64 + j] = fmaxf(s, 0.0f);
  }
  __syncthreads();
  for (int i = tid; i < 4 * Z_DIM; i += NTHREADS) {
    int bi = i >> 7, j = i & 127;
    float s = bc2[j];
#pragma unroll 8
    for (int h = 0; h < 64; ++h) s += hid[bi * 64 + h] * Wc2[h * 128 + j];
    s = tanhf(s);
#pragma unroll
    for (int sr = 0; sr < S_DIM; ++sr) zbuf[(bi * S_DIM + sr) * Z_DIM + j] = s;
  }
  __syncthreads();

  const float dtc = 0.05f, sdt = 0.22360679774997896f;
  float kld_acc = 0.0f, recon_acc = 0.0f;

  for (int t = 0; t < T_STEPS; ++t) {
    const float* h_t = path_h + (size_t)t * B_DIM * R_DIM;
    const float* hp_t = path_hpos + (size_t)t * B_DIM * R_DIM;

    if (t + 1 < T_STEPS && tid < 128) { // warm next step's h/hpos rows into cache
      __builtin_prefetch(path_h + ((size_t)(t + 1) * B_DIM + b0) * R_DIM + tid * 8, 0, 0);
      __builtin_prefetch(path_hpos + ((size_t)(t + 1) * B_DIM + b0) * R_DIM + tid * 8, 0, 0);
    }

    // P1: stage act. rows 0..31 h | rows 32..63 h+hpos ; both z halves = bf16(z)
    for (int i = tid; i < MT * R_DIM; i += NTHREADS) {
      int m = i >> 8, k = i & 255;
      int b = b0 + (m >> 3);
      float hv = h_t[(size_t)b * R_DIM + k];
      float hpv = hp_t[(size_t)b * R_DIM + k];
      act[m * ACTS + k] = (__bf16)hv;
      act[(MT + m) * ACTS + k] = (__bf16)(hv + hpv);
    }
    for (int i = tid; i < MT * Z_DIM; i += NTHREADS) {
      int m = i >> 7, j = i & 127;
      __bf16 zb = (__bf16)zbuf[i];
      act[m * ACTS + R_DIM + j] = zb;
      act[(MT + m) * ACTS + R_DIM + j] = zb;
    }
    __syncthreads();

    // P2: a1[64,512] = relu(act @ Wd1 + bd1)    (prior+poster fused)
#pragma unroll 1
    for (int i = 0; i < 4; ++i) {
      int nt = wave + 8 * i;
      f32x8 acc[4];
      gemm_multi<4>(act, ACTS, KD, Wd1s, 32, nt, bd1, lane, acc);
      int n = (nt << 4) + (lane & 15), half = lane >> 4;
#pragma unroll
      for (int mt = 0; mt < 4; ++mt)
#pragma unroll
        for (int r = 0; r < 8; ++r)
          a1[(mt * 16 + r + half * 8) * A1S + n] = (__bf16)fmaxf(acc[mt][r], 0.0f);
    }
    __syncthreads();

    // P3: ddb[64,128] = 0.1*tanh(a1 @ Wd2 + bd2)   (rows<32 prior, rows>=32 poster)
    {
      f32x8 acc[4];
      gemm_multi<4>(a1, A1S, H_DIM, Wd2s, 8, wave, bd2, lane, acc);
      int n = (wave << 4) + (lane & 15), half = lane >> 4;
#pragma unroll
      for (int mt = 0; mt < 4; ++mt)
#pragma unroll
        for (int r = 0; r < 8; ++r)
          ddb[(mt * 16 + r + half * 8) * Z_DIM + n] = 0.1f * tanhf(acc[mt][r]);
    }
    __syncthreads();

    // P4: a1[0..31] = relu(act[0..31, :256] @ Wf1 + bf1)
#pragma unroll 1
    for (int i = 0; i < 4; ++i) {
      int nt = wave + 8 * i;
      f32x8 acc[2];
      gemm_multi<2>(act, ACTS, R_DIM, Wf1s, 32, nt, bf1, lane, acc);
      int n = (nt << 4) + (lane & 15), half = lane >> 4;
#pragma unroll
      for (int mt = 0; mt < 2; ++mt)
#pragma unroll
        for (int r = 0; r < 8; ++r)
          a1[(mt * 16 + r + half * 8) * A1S + n] = (__bf16)fmaxf(acc[mt][r], 0.0f);
    }
    __syncthreads();

    // P5: diffusion = exp(a1[0..31] @ Wf2 + bf2)
    {
      f32x8 acc[2];
      gemm_multi<2>(a1, A1S, H_DIM, Wf2s, 8, wave, bf2, lane, acc);
      int n = (wave << 4) + (lane & 15), half = lane >> 4;
#pragma unroll
      for (int mt = 0; mt < 2; ++mt)
#pragma unroll
        for (int r = 0; r < 8; ++r)
          diffb[(mt * 16 + r + half * 8) * Z_DIM + n] = expf(acc[mt][r]);
    }
    __syncthreads();

    // P6: z += dt*poster + sqrt(dt)*diff*eps ; kld += ((prior-poster)/diff)^2
    //     and restage new z into act rows 0..31 for the p-net
    for (int i = tid; i < MT * Z_DIM; i += NTHREADS) {
      int m = i >> 7, j = i & 127;
      int gm = m0 + m, b = gm >> 3, s = gm & 7;
      float pr = ddb[i];
      float po = ddb[MT * Z_DIM + i];
      float di = diffb[i];
      float e = (pr - po) / di;
      kld_acc += e * e;
      float eps = noise[(((size_t)t * B_DIM + b) * S_DIM + s) * Z_DIM + j];
      zbuf[i] += dtc * po + sdt * di * eps;
      act[m * ACTS + R_DIM + j] = (__bf16)zbuf[i];
    }
    __syncthreads();

    // P7: a1[0..31] = relu(act[0..31] @ Wp1 + bp1)
#pragma unroll 1
    for (int i = 0; i < 4; ++i) {
      int nt = wave + 8 * i;
      f32x8 acc[2];
      gemm_multi<2>(act, ACTS, KD, Wp1s, 32, nt, bp1, lane, acc);
      int n = (nt << 4) + (lane & 15), half = lane >> 4;
#pragma unroll
      for (int mt = 0; mt < 2; ++mt)
#pragma unroll
        for (int r = 0; r < 8; ++r)
          a1[(mt * 16 + r + half * 8) * A1S + n] = (__bf16)fmaxf(acc[mt][r], 0.0f);
    }
    __syncthreads();

    // P8: p = a1 @ Wp2 + bp2   ([mean|logvar], 128 cols)
    {
      f32x8 acc[2];
      gemm_multi<2>(a1, A1S, H_DIM, Wp2s, 8, wave, bp2, lane, acc);
      int n = (wave << 4) + (lane & 15), half = lane >> 4;
#pragma unroll
      for (int mt = 0; mt < 2; ++mt)
#pragma unroll
        for (int r = 0; r < 8; ++r)
          pbuf[(mt * 16 + r + half * 8) * Z_DIM + n] = acc[mt][r];
    }
    __syncthreads();

    // P9: Mt = mean_d M[t,b,:]; recon += Mt * nll / (B*S)
    {
      int bi = tid >> 6, d = tid & 63;  // 4 x 64 = 256 threads
      red[tid] = Mmask[((size_t)t * B_DIM + b0 + bi) * DIN + d];
    }
    __syncthreads();
    if (tid < 4) {
      float s = 0.0f;
      for (int d = 0; d < DIN; ++d) s += red[tid * 64 + d];
      MtS[tid] = s * (1.0f / DIN);
    }
    __syncthreads();
    for (int i = tid; i < MT * DIN; i += NTHREADS) {
      int m = i >> 6, d = i & 63;
      int b = b0 + (m >> 3);
      float mean = pbuf[m * Z_DIM + d];
      float logv = pbuf[m * Z_DIM + DIN + d];
      float dv = X[((size_t)t * B_DIM + b) * DIN + d] - mean;
      float nll = 0.5f * (LOG2PI + logv + dv * dv * expf(-logv));
      recon_acc += MtS[m >> 3] * nll;
    }
    __syncthreads();
  }

  // ---- write z and reduce scalar losses ----
  for (int i = tid; i < MT * Z_DIM; i += NTHREADS)
    out[(size_t)m0 * Z_DIM + i] = zbuf[i];

  red[tid] = kld_acc;
  __syncthreads();
  for (int off = NTHREADS / 2; off > 0; off >>= 1) {
    if (tid < off) red[tid] += red[tid + off];
    __syncthreads();
  }
  if (tid == 0) atomicAdd(out + ZOUT + 2, red[0] * (0.5f * 0.05f / (float)(B_DIM * S_DIM)));
  __syncthreads();
  red[tid] = recon_acc;
  __syncthreads();
  for (int off = NTHREADS / 2; off > 0; off >>= 1) {
    if (tid < off) red[tid] += red[tid + off];
    __syncthreads();
  }
  if (tid == 0) atomicAdd(out + ZOUT + 1, red[0] * (1.0f / (float)(B_DIM * S_DIM)));
}

// ---------------- host entry ----------------
extern "C" void kernel_launch(void* const* d_in, const int* in_sizes, int n_in,
                              void* d_out, int out_size, void* d_ws, size_t ws_size,
                              hipStream_t stream) {
  const float* X    = (const float*)d_in[0];
  const float* Mm   = (const float*)d_in[1];
  const float* cov  = (const float*)d_in[2];
  const float* ph   = (const float*)d_in[3];
  const float* php  = (const float*)d_in[4];
  const float* noi  = (const float*)d_in[5];
  const float* Wc1  = (const float*)d_in[6];
  const float* bc1  = (const float*)d_in[7];
  const float* Wc2  = (const float*)d_in[8];
  const float* bc2  = (const float*)d_in[9];
  const float* Wd1  = (const float*)d_in[10];
  const float* bd1  = (const float*)d_in[11];
  const float* Wd2  = (const float*)d_in[12];
  const float* bd2  = (const float*)d_in[13];
  const float* Wf1  = (const float*)d_in[14];
  const float* bf1  = (const float*)d_in[15];
  const float* Wf2  = (const float*)d_in[16];
  const float* bf2  = (const float*)d_in[17];
  const float* Wp1  = (const float*)d_in[18];
  const float* bp1  = (const float*)d_in[19];
  const float* Wp2  = (const float*)d_in[20];
  const float* bp2  = (const float*)d_in[21];
  __bf16* ws = (__bf16*)d_ws;
  float* out = (float*)d_out;

  auto sw = [&](const float* W, __bf16* dst, int K, int N) {
    int total = K * N;
    swizzle_weights<<<(total + 255) / 256, 256, 0, stream>>>(W, dst, K, N);
  };
  sw(Wd1, ws + WD1_OFF, KD, H_DIM);
  sw(Wd2, ws + WD2_OFF, H_DIM, Z_DIM);
  sw(Wf1, ws + WF1_OFF, R_DIM, H_DIM);
  sw(Wf2, ws + WF2_OFF, H_DIM, Z_DIM);
  sw(Wp1, ws + WP1_OFF, KD, H_DIM);
  sw(Wp2, ws + WP2_OFF, H_DIM, Z_DIM);

  zero_scalars<<<1, 32, 0, stream>>>(out);

  vsdn_kernel<<<NWG, NTHREADS, 0, stream>>>(
      X, Mm, cov, ph, php, noi, Wc1, bc1, Wc2, bc2,
      bd1, bd2, bf1, bf2, bp1, bp2, ws, out);

  finalize_total<<<1, 32, 0, stream>>>(out);
}
